// LindeBuzoGrayAlgorithm_80058190397777
// MI455X (gfx1250) — matrix operand order repaired and checked by reading, not compiled, not run on hardware
//
#include <hip/hip_runtime.h>
#include <hip/hip_bf16.h>
#include <math.h>

// ---------------------------------------------------------------------------
// LBG vector quantization for MI455X (gfx1250).
// Assignment GEMM runs on the f16 WMMA pipe (v_wmma_f32_16x16x32_f16),
// DIM=24 padded to K=32 so each 16x16 tile is exactly one WMMA.
// Each wave carries TWO A fragments (32 points) so every B tile fetch feeds
// two independent WMMAs (covers the WMMA->VALU hazard window with real work).
// x (8MB as f16) is L2-resident across all Lloyd iterations.
// ---------------------------------------------------------------------------

typedef __attribute__((ext_vector_type(16))) _Float16 v16h;
typedef __attribute__((ext_vector_type(8)))  _Float16 v8h;
typedef __attribute__((ext_vector_type(8)))  float    v8f;

#define DIM      24
#define DIMP     32
#define KMAX     256
#define PERTURB  1e-5f
#define EPSC     1e-5f

// Deterministic hash-based standard normal (Box-Muller). Stands in for
// jax.random.normal(fold_in(key,ctr)) -- exact values are not checkable here.
__device__ __forceinline__ float hashnorm(unsigned a, unsigned b) {
    unsigned s = a * 0x9E3779B9u + b * 0x85EBCA6Bu + 0x7F4A7C15u;
    s ^= s >> 16; s *= 0x7FEB352Du; s ^= s >> 15; s *= 0x846CA68Bu; s ^= s >> 16;
    unsigned s2 = s ^ 0x68E31DA4u;
    s2 *= 0xB5297A4Du; s2 ^= s2 >> 13; s2 *= 0x1B56C4E9u; s2 ^= s2 >> 17;
    float u1 = ((float)((s >> 8) + 1u)) * (1.0f / 16777217.0f);   // (0,1]
    float u2 = ((float)(s2 >> 8)) * (1.0f / 16777216.0f);         // [0,1)
    return sqrtf(fmaxf(-2.0f * logf(u1), 0.0f)) * cosf(6.28318530718f * u2);
}

// --------------------------- init / prep kernels ---------------------------

__global__ void lbg_init(float* macc, float* state) {
    int t = threadIdx.x;
    if (t < DIM) macc[t] = 0.0f;
    if (t == 0) { state[0] = INFINITY; state[1] = INFINITY; state[2] = 0.0f; }
}

// x -> f16 padded rows + per-point squared norm
__global__ void lbg_xh_prep(const float* __restrict__ x, _Float16* __restrict__ xh,
                            float* __restrict__ xnorm, int N) {
    int m = blockIdx.x * blockDim.x + threadIdx.x;
    if (m >= N) return;
    float s = 0.0f;
    const float* row = x + (size_t)m * DIM;
    _Float16* orow = xh + (size_t)m * DIMP;
    #pragma unroll
    for (int d = 0; d < DIMP; ++d) {
        float v = (d < DIM) ? row[d] : 0.0f;
        orow[d] = (_Float16)v;
        if (d < DIM) s += v * v;
    }
    xnorm[m] = s;
}

__global__ void lbg_mean_accum(const float* __restrict__ x, float* __restrict__ macc, int N) {
    __shared__ float s[DIM];
    if (threadIdx.x < DIM) s[threadIdx.x] = 0.0f;
    __syncthreads();
    int m = blockIdx.x * blockDim.x + threadIdx.x;
    if (m < N) {
        const float* row = x + (size_t)m * DIM;
        #pragma unroll
        for (int d = 0; d < DIM; ++d) atomicAdd(&s[d], row[d]);
    }
    __syncthreads();
    if (threadIdx.x < DIM) atomicAdd(&macc[threadIdx.x], s[threadIdx.x]);
}

__global__ void lbg_mean_fin(const float* macc, float* cb, int N) {
    int t = threadIdx.x;
    if (t < DIM) cb[t] = macc[t] / (float)N;
}

// Zero per-iteration accumulators: cnt[256], sums[256*24], dacc[1]
__global__ void lbg_prep_iter(float* cnt, float* sums, float* dacc) {
    int i = blockIdx.x * blockDim.x + threadIdx.x;
    if (i < KMAX) cnt[i] = 0.0f;
    if (i < KMAX * DIM) sums[i] = 0.0f;
    if (i == 0) dacc[0] = 0.0f;
}

// codebook -> f16 padded rows + f32 norms; pad slots get +huge norm
__global__ void lbg_cbh_prep(const float* __restrict__ cb, _Float16* __restrict__ cbh,
                             float* __restrict__ cnorm, int K, int Kpad) {
    int n = blockIdx.x * blockDim.x + threadIdx.x;
    if (n >= Kpad) return;
    float s = 0.0f;
    _Float16* orow = cbh + (size_t)n * DIMP;
    #pragma unroll
    for (int d = 0; d < DIMP; ++d) {
        float v = (n < K && d < DIM) ? cb[n * DIM + d] : 0.0f;
        _Float16 h = (_Float16)v;
        orow[d] = h;
        float hv = (float)h;
        if (d < DIM) s += hv * hv;          // norm of the f16-rounded centroid
    }
    cnorm[n] = (n < K) ? s : 3.0e38f;        // pad slots never win argmin
}

// ----------------------- WMMA assignment (the hot loop) --------------------
// One wave per 32-point tile (two 16-row A fragments held in registers).
// Each 16-centroid B tile feeds two independent v_wmma_f32_16x16x32_f16.
__global__ __launch_bounds__(256)
void lbg_assign(const _Float16* __restrict__ xh, const float* __restrict__ xnorm,
                const _Float16* __restrict__ cbh, const float* __restrict__ cnorm,
                int N, int Kpad, int* __restrict__ idx, float* __restrict__ idxf,
                float* __restrict__ dacc) {
    const int lane = threadIdx.x & 31;
    const int wave = threadIdx.x >> 5;
    const int m0 = (blockIdx.x * 8 + wave) * 32;
    if (m0 >= N) return;                 // wave-uniform: EXEC stays all-ones

    const int half = lane >> 4;          // 0: lanes 0-15, 1: lanes 16-31
    const int l16  = lane & 15;

    // A fragments (16x32 f16 each): lane holds row m, K = half*8 + [0..7]
    // and 16 + half*8 + [0..7]  (per ISA A layout)
    const _Float16* arow0 = xh + (size_t)(m0 + l16) * DIMP;
    const _Float16* arow1 = xh + (size_t)(m0 + 16 + l16) * DIMP;
    v8h a0lo = *(const v8h*)(arow0 + half * 8);
    v8h a0hi = *(const v8h*)(arow0 + 16 + half * 8);
    v8h a1lo = *(const v8h*)(arow1 + half * 8);
    v8h a1hi = *(const v8h*)(arow1 + 16 + half * 8);
    v16h a0 = __builtin_shufflevector(a0lo, a0hi,
              0, 1, 2, 3, 4, 5, 6, 7, 8, 9, 10, 11, 12, 13, 14, 15);
    v16h a1 = __builtin_shufflevector(a1lo, a1hi,
              0, 1, 2, 3, 4, 5, 6, 7, 8, 9, 10, 11, 12, 13, 14, 15);

    float bestv0[8], bestv1[8];
    int   besti0[8], besti1[8];
    #pragma unroll
    for (int r = 0; r < 8; ++r) {
        bestv0[r] = 3.4e38f; besti0[r] = 0;
        bestv1[r] = 3.4e38f; besti1[r] = 0;
    }

    for (int n0 = 0; n0 < Kpad; n0 += 16) {
        // B fragment (32x16 f16): lane holds col n = n0+l16, K = half*8 + [0..7] and 16 + half*8 + [0..7]
        const _Float16* brow = cbh + (size_t)(n0 + l16) * DIMP + half * 8;
        v8h blo = *(const v8h*)(brow);
        v8h bhi = *(const v8h*)(brow + 16);
        v16h b = __builtin_shufflevector(blo, bhi,
                 0, 1, 2, 3, 4, 5, 6, 7, 8, 9, 10, 11, 12, 13, 14, 15);

        v8f c0 = {};
        v8f c1 = {};
        c0 = __builtin_amdgcn_wmma_f32_16x16x32_f16(
                 false, a0, false, b, (short)0, c0, false, false);
        c1 = __builtin_amdgcn_wmma_f32_16x16x32_f16(
                 false, a1, false, b, (short)0, c1, false, false);

        const int   n  = n0 + l16;
        const float cn = cnorm[n];
        // D layout: VGPR r -> row mtile + r + half*8, col n0 + l16
        #pragma unroll
        for (int r = 0; r < 8; ++r) {
            float v0 = cn - 2.0f * c0[r];        // |c|^2 - 2 x.c  (rank == dist)
            float v1 = cn - 2.0f * c1[r];
            if (v0 < bestv0[r]) { bestv0[r] = v0; besti0[r] = n; }
            if (v1 < bestv1[r]) { bestv1[r] = v1; besti1[r] = n; }
        }
    }

    // min-reduce across the 16 columns (stay within each half-group)
    #pragma unroll
    for (int off = 8; off >= 1; off >>= 1) {
        #pragma unroll
        for (int r = 0; r < 8; ++r) {
            float ov0 = __shfl_xor(bestv0[r], off, 32);
            int   oi0 = __shfl_xor(besti0[r], off, 32);
            if (ov0 < bestv0[r] || (ov0 == bestv0[r] && oi0 < besti0[r])) {
                bestv0[r] = ov0; besti0[r] = oi0;
            }
            float ov1 = __shfl_xor(bestv1[r], off, 32);
            int   oi1 = __shfl_xor(besti1[r], off, 32);
            if (ov1 < bestv1[r] || (ov1 == bestv1[r] && oi1 < besti1[r])) {
                bestv1[r] = ov1; besti1[r] = oi1;
            }
        }
    }

    if (l16 == 0) {
        float local = 0.0f;
        #pragma unroll
        for (int r = 0; r < 8; ++r) {
            int m = m0 + r + half * 8;
            idx[m]  = besti0[r];
            idxf[m] = (float)besti0[r];
            local  += fmaxf(bestv0[r] + xnorm[m], 0.0f);   // |x-c|^2
            int m1 = m0 + 16 + r + half * 8;
            idx[m1]  = besti1[r];
            idxf[m1] = (float)besti1[r];
            local   += fmaxf(bestv1[r] + xnorm[m1], 0.0f);
        }
        atomicAdd(dacc, local);
    }
}

// ------------------------- centroid scatter (LDS agg) ----------------------
__global__ __launch_bounds__(256)
void lbg_scatter(const float* __restrict__ x, const int* __restrict__ idx,
                 float* __restrict__ cnt, float* __restrict__ sums, int N, int K) {
    __shared__ float scnt[KMAX];
    __shared__ float ssum[KMAX * DIM];
    for (int i = threadIdx.x; i < K; i += blockDim.x) scnt[i] = 0.0f;
    for (int i = threadIdx.x; i < K * DIM; i += blockDim.x) ssum[i] = 0.0f;
    __syncthreads();
    int m = blockIdx.x * blockDim.x + threadIdx.x;
    if (m < N) {
        int c = idx[m];
        atomicAdd(&scnt[c], 1.0f);
        const float* row = x + (size_t)m * DIM;
        #pragma unroll
        for (int d = 0; d < DIM; ++d) atomicAdd(&ssum[c * DIM + d], row[d]);
    }
    __syncthreads();
    for (int i = threadIdx.x; i < K; i += blockDim.x)
        if (scnt[i] != 0.0f) atomicAdd(&cnt[i], scnt[i]);
    for (int i = threadIdx.x; i < K * DIM; i += blockDim.x)
        if (ssum[i] != 0.0f) atomicAdd(&sums[i], ssum[i]);
}

// ------------------- centroid finalize + convergence mask ------------------
__global__ __launch_bounds__(256)
void lbg_finalize(float* __restrict__ cb, const float* __restrict__ cnt,
                  const float* __restrict__ sums, float* __restrict__ state,
                  const float* __restrict__ dacc, int K, int it, int ctr, int N) {
    __shared__ float sval[256];
    __shared__ int   sidx[256];
    __shared__ int   s_midx;
    __shared__ int   s_write;
    __shared__ float s_cnte;
    int t = threadIdx.x;
    sval[t] = (t < K) ? cnt[t] : -1.0f;
    sidx[t] = t;
    __syncthreads();
    for (int off = 128; off >= 1; off >>= 1) {
        if (t < off) {
            float ov = sval[t + off]; int oi = sidx[t + off];
            if (ov > sval[t] || (ov == sval[t] && oi < sidx[t])) {
                sval[t] = ov; sidx[t] = oi;
            }
        }
        __syncthreads();
    }
    if (t == 0) {
        s_midx = sidx[0];
        float d    = dacc[0] / (float)N;
        float prev = state[1];
        bool  done = state[2] != 0.0f;
        bool  newly = (it > 0) && (fabsf(prev - d) < EPSC * (d + 1e-16f));
        if (!done) { state[0] = d; state[1] = d; }
        s_write = (!done && !newly) ? 1 : 0;
        if (done || newly) state[2] = 1.0f;
        int ce = 0;
        for (int i = 0; i < K; ++i) if (cnt[i] < 1.0f) ++ce;
        s_cnte = (float)ce;
    }
    __syncthreads();
    if (!s_write) return;
    int midx = s_midx;
    // 1: populated clusters -> mean
    if (t < K && cnt[t] >= 1.0f) {
        float nd = cnt[t];
        for (int d = 0; d < DIM; ++d) cb[t * DIM + d] = sums[t * DIM + d] / nd;
    }
    __syncthreads();
    // 2: empty clusters split off the most populated one
    if (t < K && cnt[t] < 1.0f) {
        for (int d = 0; d < DIM; ++d)
            cb[t * DIM + d] = cb[midx * DIM + d] - PERTURB * hashnorm((unsigned)ctr, (unsigned)(t * DIM + d));
    }
    __syncthreads();
    // 3: compensate the donor centroid
    if (t < DIM && s_cnte > 0.0f) {
        float acc = 0.0f;
        for (int i = 0; i < K; ++i)
            if (cnt[i] < 1.0f) acc += PERTURB * hashnorm((unsigned)ctr, (unsigned)(i * DIM + t));
        cb[midx * DIM + t] += acc / s_cnte;
    }
}

// ------------------------------ split kernel -------------------------------
__global__ void lbg_split(float* cb, float* state, int khalf, int ctr) {
    int t = blockIdx.x * blockDim.x + threadIdx.x;
    if (t == 0) { state[1] = state[0]; state[2] = 0.0f; }
    if (t < khalf) {
        #pragma unroll
        for (int d = 0; d < DIM; ++d) {
            float v = cb[t * DIM + d];
            float r = PERTURB * hashnorm((unsigned)ctr, (unsigned)(t * DIM + d));
            cb[t * DIM + d]            = v + r;
            cb[(t + khalf) * DIM + d]  = v - r;
        }
    }
}

__global__ void lbg_copy_out(const float* cb, const float* state, float* out, int N) {
    int t = blockIdx.x * blockDim.x + threadIdx.x;
    if (t < KMAX * DIM) out[t] = cb[t];
    if (t == KMAX * DIM) out[KMAX * DIM + N] = state[0];   // distortion scalar
}

// ------------------------------- launcher ----------------------------------
extern "C" void kernel_launch(void* const* d_in, const int* in_sizes, int n_in,
                              void* d_out, int out_size, void* d_ws, size_t ws_size,
                              hipStream_t stream) {
    const float* x = (const float*)d_in[0];
    const int N = in_sizes[0] / DIM;          // 131072
    float* out = (float*)d_out;

    // carve workspace (256B aligned slices)
    char* ws = (char*)d_ws;
    size_t off = 0;
    auto carve = [&](size_t bytes) -> void* {
        void* p = ws + off;
        off = (off + bytes + 255) & ~(size_t)255;
        return p;
    };
    _Float16* xh    = (_Float16*)carve((size_t)N * DIMP * sizeof(_Float16));
    float*    xnorm = (float*)   carve((size_t)N * sizeof(float));
    int*      idx   = (int*)     carve((size_t)N * sizeof(int));
    float*    cb    = (float*)   carve(KMAX * DIM * sizeof(float));
    _Float16* cbh   = (_Float16*)carve(KMAX * DIMP * sizeof(_Float16));
    float*    cnorm = (float*)   carve(KMAX * sizeof(float));
    float*    cnt   = (float*)   carve(KMAX * sizeof(float));
    float*    sums  = (float*)   carve(KMAX * DIM * sizeof(float));
    float*    macc  = (float*)   carve(DIM * sizeof(float));
    float*    dacc  = (float*)   carve(sizeof(float));
    float*    state = (float*)   carve(4 * sizeof(float));
    (void)ws_size; (void)n_in; (void)out_size;

    float* idxf = out + KMAX * DIM;           // idx region of output

    const int nblk256 = (N + 255) / 256;
    const int nblk_assign = (N + 255) / 256;  // 32 points/wave * 8 waves = 256/block

    lbg_init<<<1, 64, 0, stream>>>(macc, state);
    lbg_xh_prep<<<nblk256, 256, 0, stream>>>(x, xh, xnorm, N);
    lbg_mean_accum<<<nblk256, 256, 0, stream>>>(x, macc, N);
    lbg_mean_fin<<<1, 32, 0, stream>>>(macc, cb, N);

    int ctr = 0;
    int k = 1;
    while (2 * k <= KMAX) {
        lbg_split<<<1, 256, 0, stream>>>(cb, state, k, ctr); ++ctr;
        k *= 2;
        const int Kpad = (k + 15) & ~15;
        for (int it = 0; it < 6; ++it) {
            lbg_prep_iter<<<(KMAX * DIM + 255) / 256, 256, 0, stream>>>(cnt, sums, dacc);
            lbg_cbh_prep<<<(Kpad + 63) / 64, 64, 0, stream>>>(cb, cbh, cnorm, k, Kpad);
            lbg_assign<<<nblk_assign, 256, 0, stream>>>(xh, xnorm, cbh, cnorm, N, Kpad,
                                                        idx, idxf, dacc);
            lbg_scatter<<<nblk256, 256, 0, stream>>>(x, idx, cnt, sums, N, k);
            lbg_finalize<<<1, 256, 0, stream>>>(cb, cnt, sums, state, dacc, k, it, ctr, N);
            ++ctr;
        }
    }

    // final assignment with the converged 256-entry codebook
    lbg_prep_iter<<<(KMAX * DIM + 255) / 256, 256, 0, stream>>>(cnt, sums, dacc);
    lbg_cbh_prep<<<(KMAX + 63) / 64, 64, 0, stream>>>(cb, cbh, cnorm, KMAX, KMAX);
    lbg_assign<<<nblk_assign, 256, 0, stream>>>(xh, xnorm, cbh, cnorm, N, KMAX, idx, idxf, dacc);
    lbg_copy_out<<<(KMAX * DIM + 256) / 256 + 1, 256, 0, stream>>>(cb, state, out, N);
}
